// MLASelfAttention_2542620639798
// MI455X (gfx1250) — matrix operand order repaired and checked by reading, not compiled, not run on hardware
//
#include <hip/hip_runtime.h>
#include <hip/hip_bf16.h>
#include <math.h>

typedef _Float16 h16;
typedef __attribute__((ext_vector_type(16))) _Float16 v16h;
typedef __attribute__((ext_vector_type(8)))  float    v8f;
typedef __attribute__((ext_vector_type(4)))  float    v4f;

#define B_SZ 2
#define T_SZ 2048
#define C_SZ 1024
#define N_HEAD 16
#define D_HEAD 64
#define D_LATENT 32

// ---- fragment assembly: two aligned 16B chunks -> v16h -------------------
union FragU { v4f f[2]; v16h h; };

static __device__ inline v16h load_frag(const h16* p0, const h16* p1) {
    FragU u;
    u.f[0] = *(const v4f*)p0;
    u.f[1] = *(const v4f*)p1;
    return u.h;
}

// ---- fp32 -> f16 cast -----------------------------------------------------
__global__ void cast_f32_f16(const float* __restrict__ in, h16* __restrict__ out, int n) {
    int i = blockIdx.x * blockDim.x + threadIdx.x;
    if (i < n) out[i] = (h16)in[i];
}

// ---- WMMA GEMM: C[M][N] = A[M][K] @ W[N][K]^T  (A,W f16 row-major, C f32) -
// 128x128 block tile, 256 threads = 8 waves (4 m-groups x 2 n-groups),
// each wave: 2 m-subtiles x 4 n-subtiles = 8 WMMA per 32-deep k-step.
// Double-buffered LDS: one barrier per k-step, next tile prefetched to regs.
__global__ void __launch_bounds__(256)
gemm_wmma_f16(const h16* __restrict__ A, const h16* __restrict__ W,
              float* __restrict__ C, int M, int N, int K) {
    __shared__ __align__(16) h16 As[2][128][40];
    __shared__ __align__(16) h16 Bs[2][128][40];

    const int tid  = threadIdx.x;
    const int wid  = tid >> 5;
    const int lane = tid & 31;
    const int half = lane >> 4;
    const int lc   = lane & 15;
    const int m0   = blockIdx.y * 128;
    const int n0   = blockIdx.x * 128;
    const int wm   = wid & 3;            // m-subtile pair base = wm*2
    const int wn   = wid >> 2;           // n-subtile quad base = wn*4

    const int lrow = tid >> 1;           // cooperative-load row 0..127
    const int lcg  = (tid & 1) * 16;     // 16 f16 (2 x 16B) per thread per tile

    v8f acc[2][4];
#pragma unroll
    for (int mi = 0; mi < 2; ++mi)
#pragma unroll
        for (int j = 0; j < 4; ++j)
            acc[mi][j] = (v8f){0.f,0.f,0.f,0.f,0.f,0.f,0.f,0.f};

    const h16* Aptr = A + (size_t)(m0 + lrow) * K + lcg;
    const h16* Wptr = W + (size_t)(n0 + lrow) * K + lcg;

    // preload k-chunk 0 into buffer 0
    {
        v4f ra0 = *(const v4f*)(Aptr);
        v4f ra1 = *(const v4f*)(Aptr + 8);
        v4f rb0 = *(const v4f*)(Wptr);
        v4f rb1 = *(const v4f*)(Wptr + 8);
        *(v4f*)&As[0][lrow][lcg]     = ra0;
        *(v4f*)&As[0][lrow][lcg + 8] = ra1;
        *(v4f*)&Bs[0][lrow][lcg]     = rb0;
        *(v4f*)&Bs[0][lrow][lcg + 8] = rb1;
    }
    __syncthreads();

    const int nk = K >> 5;               // K / 32
    for (int kk = 0; kk < nk; ++kk) {
        const int cur = kk & 1, nxt = cur ^ 1;

        // prefetch next k-chunk into registers (hidden behind the 8 WMMAs)
        v4f ra0, ra1, rb0, rb1;
        if (kk + 1 < nk) {
            const h16* ap = Aptr + (size_t)(kk + 1) * 32;
            const h16* wp = Wptr + (size_t)(kk + 1) * 32;
            ra0 = *(const v4f*)(ap);
            ra1 = *(const v4f*)(ap + 8);
            rb0 = *(const v4f*)(wp);
            rb1 = *(const v4f*)(wp + 8);
        }

        // 8 WMMA from current buffer (2 A-frags reused across 4 B-frags)
        v16h a0 = load_frag(&As[cur][(wm*2 + 0)*16 + lc][8*half],
                            &As[cur][(wm*2 + 0)*16 + lc][16 + 8*half]);
        v16h a1 = load_frag(&As[cur][(wm*2 + 1)*16 + lc][8*half],
                            &As[cur][(wm*2 + 1)*16 + lc][16 + 8*half]);
#pragma unroll
        for (int j = 0; j < 4; ++j) {
            v16h b = load_frag(&Bs[cur][(wn*4 + j)*16 + lc][8*half],
                               &Bs[cur][(wn*4 + j)*16 + lc][16 + 8*half]);
            acc[0][j] = __builtin_amdgcn_wmma_f32_16x16x32_f16(false, a0, false, b, (short)0, acc[0][j], false, false);
            acc[1][j] = __builtin_amdgcn_wmma_f32_16x16x32_f16(false, a1, false, b, (short)0, acc[1][j], false, false);
        }

        // store prefetched chunk into the other buffer, then one barrier
        if (kk + 1 < nk) {
            *(v4f*)&As[nxt][lrow][lcg]     = ra0;
            *(v4f*)&As[nxt][lrow][lcg + 8] = ra1;
            *(v4f*)&Bs[nxt][lrow][lcg]     = rb0;
            *(v4f*)&Bs[nxt][lrow][lcg + 8] = rb1;
        }
        __syncthreads();
    }

    // epilogue: C-fragment layout -> fp32 row-major
#pragma unroll
    for (int mi = 0; mi < 2; ++mi)
#pragma unroll
        for (int j = 0; j < 4; ++j)
#pragma unroll
            for (int i = 0; i < 8; ++i) {
                int row = m0 + (wm*2 + mi)*16 + i + 8*half;
                C[(size_t)row * N + n0 + (wn*4 + j)*16 + lc] = acc[mi][j][i];
            }
}

// ---- RoPE helper (interleaved-pair variant from the reference) ------------
static __device__ inline void rope_pair(float x1, float x2, int t, int i,
                                        float& o1, float& o2) {
    const float lg_over = 9.210340371976184f / 32.0f;   // ln(10000)/32
    float f0 = expf(-(float)((2*i)     & 31) * lg_over);
    float f1 = expf(-(float)((2*i + 1) & 31) * lg_over);
    float a0 = (float)t * f0, a1 = (float)t * f1;
    o1 = x1 * cosf(a0) - x2 * sinf(a0);
    o2 = x2 * cosf(a1) + x1 * sinf(a1);
}

// ---- RoPE q: fp32 [B,T,H*Dh] -> f16 [B,H,T,Dh] ----------------------------
__global__ void rope_q_kernel(const float* __restrict__ qpre, h16* __restrict__ qh) {
    int idx = blockIdx.x * blockDim.x + threadIdx.x;   // B*T*H*32
    if (idx >= B_SZ * T_SZ * N_HEAD * 32) return;
    int i = idx & 31;
    int h = (idx >> 5) & (N_HEAD - 1);
    int t = (idx >> 9) & (T_SZ - 1);
    int b = idx >> 20;

    const float* src = qpre + (size_t)(b*T_SZ + t)*C_SZ + h*D_HEAD + 2*i;
    float o1, o2;
    rope_pair(src[0], src[1], t, i, o1, o2);
    h16* dst = qh + (((size_t)(b*N_HEAD + h))*T_SZ + t)*D_HEAD + 2*i;
    dst[0] = (h16)o1; dst[1] = (h16)o2;
}

// ---- latent -> head up-projection (+RoPE on k), f16 [B,H,T,Dh] ------------
__global__ void upproj_kv_kernel(const float* __restrict__ klat, const float* __restrict__ vlat,
                                 const float* __restrict__ Wku,  const float* __restrict__ Wvu,
                                 h16* __restrict__ kh, h16* __restrict__ vh) {
    int idx = blockIdx.x * blockDim.x + threadIdx.x;   // B*T*H*32 (i = d/2)
    if (idx >= B_SZ * T_SZ * N_HEAD * 32) return;
    int i = idx & 31;
    int h = (idx >> 5) & (N_HEAD - 1);
    int t = (idx >> 9) & (T_SZ - 1);
    int b = idx >> 20;

    const float* kl = klat + (size_t)(b*T_SZ + t)*(N_HEAD*D_LATENT) + h*D_LATENT;
    const float* vl = vlat + (size_t)(b*T_SZ + t)*(N_HEAD*D_LATENT) + h*D_LATENT;
    int d0 = 2*i, d1 = 2*i + 1;
    float k0 = 0.f, k1 = 0.f, v0 = 0.f, v1 = 0.f;
#pragma unroll 8
    for (int l = 0; l < D_LATENT; ++l) {
        float klv = kl[l], vlv = vl[l];
        k0 += klv * Wku[d0*D_LATENT + l];
        k1 += klv * Wku[d1*D_LATENT + l];
        v0 += vlv * Wvu[d0*D_LATENT + l];
        v1 += vlv * Wvu[d1*D_LATENT + l];
    }
    float r0, r1;
    rope_pair(k0, k1, t, i, r0, r1);
    size_t base = (((size_t)(b*N_HEAD + h))*T_SZ + t)*D_HEAD;
    kh[base + d0] = (h16)r0; kh[base + d1] = (h16)r1;
    vh[base + d0] = (h16)v0; vh[base + d1] = (h16)v1;
}

// ---- flash attention: per block (64 q-rows, one head), 4 waves ------------
__global__ void __launch_bounds__(128)
flash_attn_kernel(const h16* __restrict__ Q, const h16* __restrict__ Kx,
                  const h16* __restrict__ Vx, h16* __restrict__ O, float scale) {
    __shared__ __align__(16) h16 Ks[32][72];     // 32 keys x 64 dims
    __shared__ __align__(16) h16 Vt[64][40];     // transposed: [dim][key]
    __shared__ __align__(16) h16 Ps[4][16][40];  // per-wave P tile 16x32

    const int tid  = threadIdx.x;
    const int wid  = tid >> 5;
    const int lane = tid & 31;
    const int half = lane >> 4;
    const int lc   = lane & 15;
    const int b    = blockIdx.z, h = blockIdx.y;
    const int q0b  = blockIdx.x * 64;
    const int q0w  = q0b + wid * 16;
    const size_t headbase = ((size_t)(b*N_HEAD + h)) * T_SZ;

    // Q fragments held in registers: 16 rows x 64 dims = 2 A-fragments
    v16h qf[2];
#pragma unroll
    for (int c = 0; c < 2; ++c) {
        const h16* qr = Q + (headbase + q0w + lc) * D_HEAD + c*32;
        qf[c] = load_frag(qr + 8*half, qr + 16 + 8*half);
    }

    v8f o0 = {0.f,0.f,0.f,0.f,0.f,0.f,0.f,0.f}, o1 = o0, o2 = o0, o3 = o0;
    float mrow[8], lrow[8];
#pragma unroll
    for (int i = 0; i < 8; ++i) { mrow[i] = -INFINITY; lrow[i] = 0.f; }

    const int ntiles = 2 * blockIdx.x + 2;      // causal bound for whole block
    for (int kt = 0; kt < ntiles; ++kt) {
        const int kk0 = kt * 32;
        __syncthreads();
        {   // cooperative staging: 128 threads, row = tid&31, chunk = tid>>5
            const int row = tid & 31, ch = tid >> 5;
            const h16* kr = Kx + (headbase + kk0 + row) * D_HEAD + ch*16;
            *(v4f*)&Ks[row][ch*16]     = *(const v4f*)(kr);
            *(v4f*)&Ks[row][ch*16 + 8] = *(const v4f*)(kr + 8);
            const h16* vr = Vx + (headbase + kk0 + row) * D_HEAD + ch*16;
#pragma unroll
            for (int dd = 0; dd < 16; ++dd) Vt[ch*16 + dd][row] = vr[dd];
        }
        __syncthreads();

        if (kk0 <= q0w + 15) {
            v8f s0 = {0.f,0.f,0.f,0.f,0.f,0.f,0.f,0.f}, s1 = s0;
#pragma unroll
            for (int c = 0; c < 2; ++c) {
                v16h bk0 = load_frag(&Ks[lc][c*32 + 8*half],      &Ks[lc][c*32 + 16 + 8*half]);
                v16h bk1 = load_frag(&Ks[16 + lc][c*32 + 8*half], &Ks[16 + lc][c*32 + 16 + 8*half]);
                s0 = __builtin_amdgcn_wmma_f32_16x16x32_f16(false, qf[c], false, bk0, (short)0, s0, false, false);
                s1 = __builtin_amdgcn_wmma_f32_16x16x32_f16(false, qf[c], false, bk1, (short)0, s1, false, false);
            }

            // scale + causal mask + online softmax (rows split over lane halves)
            float alpha[8];
#pragma unroll
            for (int i = 0; i < 8; ++i) {
                int rg = q0w + i + 8*half;
                float x0 = (kk0 + lc      <= rg) ? s0[i] * scale : -INFINITY;
                float x1 = (kk0 + 16 + lc <= rg) ? s1[i] * scale : -INFINITY;
                float rm = fmaxf(x0, x1);
#pragma unroll
                for (int off = 8; off >= 1; off >>= 1) rm = fmaxf(rm, __shfl_xor(rm, off, 16));
                float mnew = fmaxf(mrow[i], rm);
                alpha[i] = expf(mrow[i] - mnew);
                float p0 = expf(x0 - mnew);
                float p1 = expf(x1 - mnew);
                s0[i] = p0; s1[i] = p1;
                float rs = p0 + p1;
#pragma unroll
                for (int off = 8; off >= 1; off >>= 1) rs += __shfl_xor(rs, off, 16);
                lrow[i] = lrow[i] * alpha[i] + rs;
                mrow[i] = mnew;
            }
#pragma unroll
            for (int i = 0; i < 8; ++i) {
                o0[i] *= alpha[i]; o1[i] *= alpha[i];
                o2[i] *= alpha[i]; o3[i] *= alpha[i];
            }

            // re-shape P (C-fragment layout) -> A-fragment via per-wave LDS tile
#pragma unroll
            for (int i = 0; i < 8; ++i) {
                Ps[wid][i + 8*half][lc]      = (h16)s0[i];
                Ps[wid][i + 8*half][16 + lc] = (h16)s1[i];
            }
            asm volatile("s_wait_dscnt 0x0" ::: "memory");   // wave-local DS RAW fence
            v16h pa = load_frag(&Ps[wid][lc][8*half], &Ps[wid][lc][16 + 8*half]);

#pragma unroll
            for (int nt = 0; nt < 4; ++nt) {
                v16h bv = load_frag(&Vt[nt*16 + lc][8*half], &Vt[nt*16 + lc][16 + 8*half]);
                v8f* oo = (nt == 0) ? &o0 : (nt == 1) ? &o1 : (nt == 2) ? &o2 : &o3;
                *oo = __builtin_amdgcn_wmma_f32_16x16x32_f16(false, pa, false, bv, (short)0, *oo, false, false);
            }
        }
    }

    // normalize and write [B,T,H*Dh] f16 (feeds the Wc GEMM directly)
#pragma unroll
    for (int i = 0; i < 8; ++i) {
        int r = q0w + i + 8*half;
        float inv = 1.0f / lrow[i];
        size_t base = (size_t)(b*T_SZ + r) * C_SZ + h * D_HEAD;
        O[base +  0 + lc] = (h16)(o0[i] * inv);
        O[base + 16 + lc] = (h16)(o1[i] * inv);
        O[base + 32 + lc] = (h16)(o2[i] * inv);
        O[base + 48 + lc] = (h16)(o3[i] * inv);
    }
}

// ---------------------------------------------------------------------------
extern "C" void kernel_launch(void* const* d_in, const int* in_sizes, int n_in,
                              void* d_out, int out_size, void* d_ws, size_t ws_size,
                              hipStream_t stream) {
    const float* x   = (const float*)d_in[0];
    const float* Wq  = (const float*)d_in[1];
    const float* Wk  = (const float*)d_in[2];
    const float* Wv  = (const float*)d_in[3];
    const float* Wku = (const float*)d_in[4];
    const float* Wvu = (const float*)d_in[5];
    const float* Wc  = (const float*)d_in[6];

    float* y_out    = (float*)d_out;                          // 2*2048*1024
    float* klat_out = (float*)d_out + (size_t)4194304;        // 2*2048*512
    float* vlat_out = (float*)d_out + (size_t)6291456;        // 2*2048*512

    const size_t MT = (size_t)B_SZ * T_SZ;                    // 4096
    char* ws = (char*)d_ws;
    h16* x_h    = (h16*)ws;  ws += MT * C_SZ * sizeof(h16);
    h16* Wq_h   = (h16*)ws;  ws += (size_t)C_SZ * C_SZ * sizeof(h16);
    h16* Wk_h   = (h16*)ws;  ws += (size_t)512 * C_SZ * sizeof(h16);
    h16* Wv_h   = (h16*)ws;  ws += (size_t)512 * C_SZ * sizeof(h16);
    h16* Wc_h   = (h16*)ws;  ws += (size_t)C_SZ * C_SZ * sizeof(h16);
    float* qpre = (float*)ws; ws += MT * C_SZ * sizeof(float);
    h16* q_h    = (h16*)ws;  ws += MT * C_SZ * sizeof(h16);
    h16* k_h    = (h16*)ws;  ws += MT * C_SZ * sizeof(h16);
    h16* v_h    = (h16*)ws;  ws += MT * C_SZ * sizeof(h16);
    h16* attn_h = (h16*)ws;  ws += MT * C_SZ * sizeof(h16);

    // 1) casts
    cast_f32_f16<<<(4194304 + 255) / 256, 256, 0, stream>>>(x,  x_h,  4194304);
    cast_f32_f16<<<(1048576 + 255) / 256, 256, 0, stream>>>(Wq, Wq_h, 1048576);
    cast_f32_f16<<<( 524288 + 255) / 256, 256, 0, stream>>>(Wk, Wk_h,  524288);
    cast_f32_f16<<<( 524288 + 255) / 256, 256, 0, stream>>>(Wv, Wv_h,  524288);
    cast_f32_f16<<<(1048576 + 255) / 256, 256, 0, stream>>>(Wc, Wc_h, 1048576);

    // 2) projections (WMMA, 128x128 tiles)
    dim3 gq(C_SZ / 128, (int)MT / 128);
    dim3 gl(512  / 128, (int)MT / 128);
    gemm_wmma_f16<<<gq, 256, 0, stream>>>(x_h, Wq_h, qpre,     (int)MT, C_SZ, C_SZ);
    gemm_wmma_f16<<<gl, 256, 0, stream>>>(x_h, Wk_h, klat_out, (int)MT, 512,  C_SZ);
    gemm_wmma_f16<<<gl, 256, 0, stream>>>(x_h, Wv_h, vlat_out, (int)MT, 512,  C_SZ);

    // 3) RoPE(q) and latent up-projection (+RoPE on k)
    int nsmall = B_SZ * T_SZ * N_HEAD * 32;
    rope_q_kernel<<<(nsmall + 255) / 256, 256, 0, stream>>>(qpre, q_h);
    upproj_kv_kernel<<<(nsmall + 255) / 256, 256, 0, stream>>>(klat_out, vlat_out, Wku, Wvu, k_h, v_h);

    // 4) flash attention (WMMA), scale = 1/sqrt(64)
    flash_attn_kernel<<<dim3(T_SZ / 64, N_HEAD, B_SZ), 128, 0, stream>>>(q_h, k_h, v_h, attn_h, 0.125f);

    // 5) output projection (WMMA, 128x128 tiles)
    gemm_wmma_f16<<<gq, 256, 0, stream>>>(attn_h, Wc_h, y_out, (int)MT, C_SZ, C_SZ);
}
